// SelfAttn_31542239822548
// MI455X (gfx1250) — matrix operand order repaired and checked by reading, not compiled, not run on hardware
//
#include <hip/hip_runtime.h>
#include <hip/hip_bf16.h>

// ---------- model dims ----------
#define BATCH   2
#define C_IN    256
#define HH      48
#define HW      2304      // 48*48
#define HEADS   8
#define DIM     64
#define DMODEL  512       // HEADS*DIM
#define OUTC    256
#define BC      128       // flash attention key-block width

typedef __attribute__((ext_vector_type(16))) __bf16        v16bf;
typedef __attribute__((ext_vector_type(8)))  float         v8f;
typedef __attribute__((ext_vector_type(8)))  unsigned short ushort8;

union BF16x16 {
  v16bf   v;
  ushort8 h[2];
};

__device__ __forceinline__ unsigned short f2bf(float f) {
  unsigned int u = __builtin_bit_cast(unsigned int, f);
  return (unsigned short)((u + 0x7FFFu + ((u >> 16) & 1u)) >> 16);
}

// 64-byte global->LDS async copy (4x b128); INST_OFFSET advances both the LDS
// and the global address, so one address pair covers the whole strip.
__device__ __forceinline__ void async_copy64(unsigned lds_off, const void* g) {
  asm volatile(
      "global_load_async_to_lds_b128 %0, %1, off\n\t"
      "global_load_async_to_lds_b128 %0, %1, off offset:16\n\t"
      "global_load_async_to_lds_b128 %0, %1, off offset:32\n\t"
      "global_load_async_to_lds_b128 %0, %1, off offset:48"
      :: "v"(lds_off), "v"(g) : "memory");
}

__device__ __forceinline__ void wait_async_all() {
  asm volatile("s_wait_asynccnt 0x0" ::: "memory");
}

// ---------------------------------------------------------------------------
// Kernel: f32 -> bf16 weight conversion
// ---------------------------------------------------------------------------
__global__ __launch_bounds__(256) void cvt_bf16_kernel(
    const float* __restrict__ in, unsigned short* __restrict__ out, int n) {
  int i = blockIdx.x * 256 + threadIdx.x;
  if (i < n) out[i] = f2bf(in[i]);
}

// ---------------------------------------------------------------------------
// Kernel: 3x depthwise 7x7 conv (+bias), output bf16 in [b][pix][c] layout
// grid = BATCH*C_IN blocks (one channel plane each), 256 threads
// ---------------------------------------------------------------------------
__global__ __launch_bounds__(256) void dwconv_kernel(
    const float* __restrict__ x,
    const float* __restrict__ wq, const float* __restrict__ bq,
    const float* __restrict__ wk, const float* __restrict__ bk,
    const float* __restrict__ wv, const float* __restrict__ bv,
    unsigned short* __restrict__ yq,
    unsigned short* __restrict__ yk,
    unsigned short* __restrict__ yv) {
  __shared__ float tile[54 * 54];
  const int b = blockIdx.x >> 8;
  const int c = blockIdx.x & 255;
  const float* xp = x + ((size_t)(b * C_IN + c)) * HW;

  for (int i = threadIdx.x; i < 54 * 54; i += 256) {
    int ty = i / 54, tx = i % 54;
    int sy = ty - 3, sx = tx - 3;
    float val = 0.f;
    if (sy >= 0 && sy < HH && sx >= 0 && sx < HH) val = xp[sy * HH + sx];
    tile[i] = val;
  }
  __syncthreads();

  const float bqv = bq[c], bkv = bk[c], bvv = bv[c];
  const float* wqc = wq + c * 49;
  const float* wkc = wk + c * 49;
  const float* wvc = wv + c * 49;

  for (int p = threadIdx.x; p < HW; p += 256) {
    int oy = p / HH, ox = p % HH;
    float aq = 0.f, ak = 0.f, av = 0.f;
#pragma unroll
    for (int ky = 0; ky < 7; ky++) {
#pragma unroll
      for (int kx = 0; kx < 7; kx++) {
        float xv = tile[(oy + ky) * 54 + (ox + kx)];
        int wi = ky * 7 + kx;
        aq = fmaf(xv, wqc[wi], aq);
        ak = fmaf(xv, wkc[wi], ak);
        av = fmaf(xv, wvc[wi], av);
      }
    }
    size_t o = ((size_t)b * HW + p) * C_IN + c;
    yq[o] = f2bf(aq + bqv);
    yk[o] = f2bf(ak + bkv);
    yv[o] = f2bf(av + bvv);
  }
}

// ---------------------------------------------------------------------------
// Kernel: generic WMMA GEMM  D[pix][col] = sum_k A[b][pix][K] * Bm[col][K] + bias
// grid.x = BATCH * (HW/128); 256 threads = 8 waves, 16 pixel rows per wave.
// layout 0: bf16 out [b*8+head][pix][64]  (q/k)
// layout 1: bf16 out [b*8+head][64][pix]  (v)
// layout 2: f32  out [b][col][pix]        (final conv)
// ---------------------------------------------------------------------------
__global__ __launch_bounds__(256) void pw_gemm_kernel(
    const unsigned short* __restrict__ A,
    const unsigned short* __restrict__ Bm,
    const float* __restrict__ bias,
    void* __restrict__ outp,
    int K, int ncol, int layout) {
  const int b    = blockIdx.x / (HW / 128);
  const int pblk = blockIdx.x % (HW / 128);
  const int wave = threadIdx.x >> 5;
  const int lane = threadIdx.x & 31;
  const int m16  = lane & 15;
  const int hh   = lane >> 4;
  const int pixbase = pblk * 128 + wave * 16;

  const v8f vzero = {0.f, 0.f, 0.f, 0.f, 0.f, 0.f, 0.f, 0.f};
  const int nkk = K >> 5;
  const unsigned short* arow = A + ((size_t)b * HW + pixbase + m16) * K;

  for (int cg = 0; cg < (ncol >> 7); cg++) {
    v8f acc[8];
#pragma unroll
    for (int ct = 0; ct < 8; ct++) acc[ct] = vzero;

    for (int kk = 0; kk < nkk; kk++) {
      BF16x16 af;
      af.h[0] = *(const ushort8*)(arow + kk * 32 + 16 * hh);
      af.h[1] = *(const ushort8*)(arow + kk * 32 + 16 * hh + 8);
#pragma unroll
      for (int ct = 0; ct < 8; ct++) {
        const unsigned short* brow =
            Bm + (size_t)(cg * 128 + ct * 16 + m16) * K + kk * 32 + 16 * hh;
        BF16x16 bf;
        bf.h[0] = *(const ushort8*)(brow);
        bf.h[1] = *(const ushort8*)(brow + 8);
        acc[ct] = __builtin_amdgcn_wmma_f32_16x16x32_bf16(
            false, af.v, false, bf.v, (short)0, acc[ct], false, false);
      }
    }

#pragma unroll
    for (int ct = 0; ct < 8; ct++) {
      int col = cg * 128 + ct * 16 + m16;
      float bv = bias[col];
#pragma unroll
      for (int r = 0; r < 8; r++) {
        int pix = pixbase + r + 8 * hh;
        float val = acc[ct][r] + bv;
        if (layout == 0) {
          int head = col >> 6, dl = col & 63;
          ((unsigned short*)outp)[(((size_t)b * HEADS + head) * HW + pix) * DIM + dl] =
              f2bf(val);
        } else if (layout == 1) {
          int head = col >> 6, dl = col & 63;
          ((unsigned short*)outp)[(((size_t)b * HEADS + head) * DIM + dl) * HW + pix] =
              f2bf(val);
        } else {
          ((float*)outp)[((size_t)b * ncol + col) * HW + pix] = val;
        }
      }
    }
  }
}

// ---------------------------------------------------------------------------
// Kernel: flash attention.  softmax(Q^T K) applied to V, fused online.
// Q,K: [bh][pix][64] bf16;  V: [bh][64][pix] bf16;  O: [b][pix][512] bf16
// grid = (HW/128, 16); 256 threads = 8 waves; 16 query rows per wave.
// Key blocks of BC=128, K/V staged with double-buffered async LDS DMA.
// ---------------------------------------------------------------------------
__global__ __launch_bounds__(256) void flash_attn_kernel(
    const unsigned short* __restrict__ Q,
    const unsigned short* __restrict__ Km,
    const unsigned short* __restrict__ V,
    unsigned short* __restrict__ O) {
  __shared__ unsigned short kbuf[2][BC * DIM];    // [key][dim]   16KB each
  __shared__ unsigned short vbuf[2][DIM * BC];    // [dim][key]   16KB each
  __shared__ unsigned short pbuf[8][16 * BC];     // per-wave P   4KB each

  const int bh   = blockIdx.y;
  const int qblk = blockIdx.x;
  const int wave = threadIdx.x >> 5;
  const int lane = threadIdx.x & 31;
  const int m16  = lane & 15;
  const int hh   = lane >> 4;
  const int qrow = qblk * 128 + wave * 16 + m16;
  const int t    = threadIdx.x;

  // Q fragments live in registers for the whole kernel (A-operand layout)
  BF16x16 qa[2];
  {
    const unsigned short* qp = Q + ((size_t)bh * HW + qrow) * DIM;
#pragma unroll
    for (int kk = 0; kk < 2; kk++) {
      qa[kk].h[0] = *(const ushort8*)(qp + kk * 32 + 16 * hh);
      qa[kk].h[1] = *(const ushort8*)(qp + kk * 32 + 16 * hh + 8);
    }
  }

  const v8f vzero = {0.f, 0.f, 0.f, 0.f, 0.f, 0.f, 0.f, 0.f};
  v8f oacc[4];
#pragma unroll
  for (int ot = 0; ot < 4; ot++) oacc[ot] = vzero;
  float mi[8], li[8];
#pragma unroll
  for (int r = 0; r < 8; r++) { mi[r] = -1e30f; li[r] = 0.f; }

  const unsigned short* kg0 = Km + (size_t)bh * HW * DIM;
  const unsigned short* vg0 = V + (size_t)bh * DIM * HW;
  const int vdim = t >> 2;            // V staging: dim row per 4 threads
  const int vkq  = (t & 3) * 32;      // 32-key (64B) chunk within the row

  auto issue = [&](int kb, int buf) {
    // K block: 128 keys x 64 dims, [key][dim]; 64B strip per thread
    const unsigned short* kg = kg0 + (size_t)kb * BC * DIM + t * 32;
    async_copy64((unsigned)(size_t)&kbuf[buf][t * 32], kg);
    // V block: [dim][key] — natural slice of v's [dim][pix] layout
    const unsigned short* vg = vg0 + (size_t)vdim * HW + kb * BC + vkq;
    async_copy64((unsigned)(size_t)&vbuf[buf][vdim * BC + vkq], vg);
  };

  issue(0, 0);

  for (int kb = 0; kb < HW / BC; kb++) {
    const int cur = kb & 1;
    wait_async_all();      // this wave's staged K/V for kb complete
    __syncthreads();       // all waves' DMA complete; prev block reads done
    if (kb + 1 < HW / BC) issue(kb + 1, cur ^ 1);  // overlap with compute

    const unsigned short* kc = &kbuf[cur][0];
    const unsigned short* vc = &vbuf[cur][0];

    // S = Q^T K : 16 query rows x 128 keys (8 column tiles), K-dim = 64
    v8f s[8];
#pragma unroll
    for (int ct = 0; ct < 8; ct++) s[ct] = vzero;
#pragma unroll
    for (int kk = 0; kk < 2; kk++) {
#pragma unroll
      for (int ct = 0; ct < 8; ct++) {
        const unsigned short* kp = kc + (ct * 16 + m16) * DIM + kk * 32 + 16 * hh;
        BF16x16 bf;
        bf.h[0] = *(const ushort8*)(kp);
        bf.h[1] = *(const ushort8*)(kp + 8);
        s[ct] = __builtin_amdgcn_wmma_f32_16x16x32_bf16(
            false, qa[kk].v, false, bf.v, (short)0, s[ct], false, false);
      }
    }

    // online softmax over the 128 keys of this block (row = query)
    unsigned short* pw = &pbuf[wave][0];
#pragma unroll
    for (int r = 0; r < 8; r++) {
      float mx = s[0][r];
#pragma unroll
      for (int ct = 1; ct < 8; ct++) mx = fmaxf(mx, s[ct][r]);
#pragma unroll
      for (int off = 1; off < 16; off <<= 1)
        mx = fmaxf(mx, __shfl_xor(mx, off, 32));
      float mnew = fmaxf(mi[r], mx);
      float sum = 0.f;
#pragma unroll
      for (int ct = 0; ct < 8; ct++) {
        float p = __expf(s[ct][r] - mnew);
        s[ct][r] = p;
        sum += p;
      }
#pragma unroll
      for (int off = 1; off < 16; off <<= 1)
        sum += __shfl_xor(sum, off, 32);
      float scale = __expf(mi[r] - mnew);
      li[r] = li[r] * scale + sum;
      mi[r] = mnew;
#pragma unroll
      for (int ot = 0; ot < 4; ot++) oacc[ot][r] *= scale;
      // stash P (C-layout -> LDS row-major [qrow][key])
#pragma unroll
      for (int ct = 0; ct < 8; ct++)
        pw[(r + 8 * hh) * BC + ct * 16 + m16] = f2bf(s[ct][r]);
    }

    // O += P @ V : A = P [16 x 128keys], B = V (stored [dim][key])
#pragma unroll
    for (int kk = 0; kk < 4; kk++) {
      BF16x16 pa;
      pa.h[0] = *(const ushort8*)(pw + m16 * BC + kk * 32 + 16 * hh);
      pa.h[1] = *(const ushort8*)(pw + m16 * BC + kk * 32 + 16 * hh + 8);
#pragma unroll
      for (int ot = 0; ot < 4; ot++) {
        const unsigned short* vp = vc + (ot * 16 + m16) * BC + kk * 32 + 16 * hh;
        BF16x16 vb;
        vb.h[0] = *(const ushort8*)(vp);
        vb.h[1] = *(const ushort8*)(vp + 8);
        oacc[ot] = __builtin_amdgcn_wmma_f32_16x16x32_bf16(
            false, pa.v, false, vb.v, (short)0, oacc[ot], false, false);
      }
    }
  }

  // epilogue: normalize and write bf16 [b][pix][512] (A-layout for final conv)
  const int b = bh >> 3, head = bh & 7;
#pragma unroll
  for (int ot = 0; ot < 4; ot++) {
#pragma unroll
    for (int r = 0; r < 8; r++) {
      float val = oacc[ot][r] / li[r];
      int row = qblk * 128 + wave * 16 + r + 8 * hh;
      int dg  = head * DIM + ot * 16 + m16;
      O[((size_t)b * HW + row) * DMODEL + dg] = f2bf(val);
    }
  }
}

// ---------------------------------------------------------------------------
extern "C" void kernel_launch(void* const* d_in, const int* in_sizes, int n_in,
                              void* d_out, int out_size, void* d_ws, size_t ws_size,
                              hipStream_t stream) {
  (void)in_sizes; (void)n_in; (void)out_size; (void)ws_size;
  const float* x     = (const float*)d_in[0];
  const float* wq_dw = (const float*)d_in[1];
  const float* bq_dw = (const float*)d_in[2];
  const float* wq_pw = (const float*)d_in[3];
  const float* bq_pw = (const float*)d_in[4];
  const float* wk_dw = (const float*)d_in[5];
  const float* bk_dw = (const float*)d_in[6];
  const float* wk_pw = (const float*)d_in[7];
  const float* bk_pw = (const float*)d_in[8];
  const float* wv_dw = (const float*)d_in[9];
  const float* bv_dw = (const float*)d_in[10];
  const float* wv_pw = (const float*)d_in[11];
  const float* bv_pw = (const float*)d_in[12];
  const float* w_out = (const float*)d_in[13];
  const float* b_out = (const float*)d_in[14];

  char* ws = (char*)d_ws;
  size_t off = 0;
  auto alloc = [&](size_t bytes) -> char* {
    char* p = ws + off;
    off += (bytes + 255) & ~(size_t)255;
    return p;
  };
  unsigned short* yq  = (unsigned short*)alloc((size_t)BATCH * HW * C_IN * 2);
  unsigned short* yk  = (unsigned short*)alloc((size_t)BATCH * HW * C_IN * 2);
  unsigned short* yv  = (unsigned short*)alloc((size_t)BATCH * HW * C_IN * 2);
  unsigned short* qb  = (unsigned short*)alloc((size_t)BATCH * HEADS * HW * DIM * 2);
  unsigned short* kb  = (unsigned short*)alloc((size_t)BATCH * HEADS * HW * DIM * 2);
  unsigned short* vb  = (unsigned short*)alloc((size_t)BATCH * HEADS * HW * DIM * 2);
  unsigned short* ao  = (unsigned short*)alloc((size_t)BATCH * HW * DMODEL * 2);
  unsigned short* wpq = (unsigned short*)alloc((size_t)DMODEL * C_IN * 2);
  unsigned short* wpk = (unsigned short*)alloc((size_t)DMODEL * C_IN * 2);
  unsigned short* wpv = (unsigned short*)alloc((size_t)DMODEL * C_IN * 2);
  unsigned short* wob = (unsigned short*)alloc((size_t)OUTC * DMODEL * 2);

  // weight conversions (f32 -> bf16), deterministic each launch
  int nw = DMODEL * C_IN;
  cvt_bf16_kernel<<<(nw + 255) / 256, 256, 0, stream>>>(wq_pw, wpq, nw);
  cvt_bf16_kernel<<<(nw + 255) / 256, 256, 0, stream>>>(wk_pw, wpk, nw);
  cvt_bf16_kernel<<<(nw + 255) / 256, 256, 0, stream>>>(wv_pw, wpv, nw);
  int no = OUTC * DMODEL;
  cvt_bf16_kernel<<<(no + 255) / 256, 256, 0, stream>>>(w_out, wob, no);

  // depthwise 7x7 for all three projections
  dwconv_kernel<<<BATCH * C_IN, 256, 0, stream>>>(
      x, wq_dw, bq_dw, wk_dw, bk_dw, wv_dw, bv_dw, yq, yk, yv);

  // pointwise 1x1 as WMMA GEMMs, scattered into attention-friendly layouts
  int gemm_blocks = BATCH * (HW / 128);
  pw_gemm_kernel<<<gemm_blocks, 256, 0, stream>>>(yq, wpq, bq_pw, qb, C_IN, DMODEL, 0);
  pw_gemm_kernel<<<gemm_blocks, 256, 0, stream>>>(yk, wpk, bk_pw, kb, C_IN, DMODEL, 0);
  pw_gemm_kernel<<<gemm_blocks, 256, 0, stream>>>(yv, wpv, bv_pw, vb, C_IN, DMODEL, 1);

  // fused flash attention (async double-buffered K/V staging)
  flash_attn_kernel<<<dim3(HW / 128, BATCH * HEADS), 256, 0, stream>>>(qb, kb, vb, ao);

  // output 1x1 conv (f32 out + bias)
  pw_gemm_kernel<<<gemm_blocks, 256, 0, stream>>>(ao, wob, b_out, d_out, DMODEL, OUTC, 2);
}
